// PagedAttentionOp_48369921688052
// MI455X (gfx1250) — compile-verified, matrix operand order checked by reading, not run on hardware
//
#include <hip/hip_runtime.h>

// Paged attention decode for MI455X (gfx1250, wave32).
// B=32, NQ=32, NKV=8 (G=4), D=128, pages: 4096 x 16 tokens, 128 pages/seq (S=2048).
// Memory-bound (~2 FLOP/byte): stream K/V pages coalesced, WMMA f16 for QK^T tiles.

typedef __attribute__((ext_vector_type(16))) _Float16 v16h;
typedef __attribute__((ext_vector_type(8)))  float    v8f;

#define B_         32
#define NQ_        32
#define NKV_       8
#define D_         128
#define G_         4          // NQ / NKV
#define NUM_PAGES_ 4096
#define TPP_       16
#define P_         128        // pages per sequence slot
#define WAVES_     8
#define THREADS_   (WAVES_ * 32)
#define MASKV      (-1.0e30f)

__device__ __forceinline__ float wave_max32(float v) {
#pragma unroll
  for (int o = 16; o > 0; o >>= 1) v = fmaxf(v, __shfl_xor(v, o, 32));
  return v;
}
__device__ __forceinline__ float wave_sum32(float v) {
#pragma unroll
  for (int o = 16; o > 0; o >>= 1) v += __shfl_xor(v, o, 32);
  return v;
}

__global__ __launch_bounds__(THREADS_) void paged_attn_decode_kernel(
    const float* __restrict__ query,        // [B,1,NQ,D]
    const float* __restrict__ key_pages,    // [NKV,NUM_PAGES,TPP,D]
    const float* __restrict__ value_pages,  // [NKV,NUM_PAGES,TPP,D]
    const int*   __restrict__ page_map,     // [B,P]
    const int*   __restrict__ seq_lengths,  // [B]
    float*       __restrict__ out)          // [B,1,NQ,D]
{
  const int b    = blockIdx.x / NKV_;
  const int kv   = blockIdx.x % NKV_;
  const int lane = threadIdx.x & 31;
  const int wv   = threadIdx.x >> 5;
  const int seq_len = seq_lengths[b];

  __shared__ float s_m[WAVES_][G_];
  __shared__ float s_l[WAVES_][G_];
  __shared__ float s_acc[WAVES_][G_][D_];

  // WMMA 16-bit A/B fragment lane pattern (CDNA5 ISA 7.12.2):
  //   lanes 0-15  -> row/col = lane,   halves = dims {0..7, 16..23} of chunk
  //   lanes 16-31 -> row/col = lane-16, halves = dims {8..15, 24..31} of chunk
  const int row16 = lane & 15;
  const int hsel  = (lane >> 4) ? 8 : 0;

  // ---- A matrix: Q fragment, M = q-head (rows >= G zero-padded), 4 chunks of 32 dims ----
  v16h qf[4];
#pragma unroll
  for (int c = 0; c < 4; ++c)
#pragma unroll
    for (int j = 0; j < 16; ++j) qf[c][j] = (_Float16)0.0f;
  if (row16 < G_) {
    const float* qrow = query + ((size_t)b * NQ_ + kv * G_ + row16) * D_;
#pragma unroll
    for (int c = 0; c < 4; ++c) {
      const int base = c * 32 + hsel;
#pragma unroll
      for (int j = 0; j < 8; ++j) qf[c][j]     = (_Float16)qrow[base + j];
#pragma unroll
      for (int j = 0; j < 8; ++j) qf[c][8 + j] = (_Float16)qrow[base + 16 + j];
    }
  }

  // ---- flash-attention running state; each lane owns output dims [4*lane, 4*lane+4) ----
  const int doff = lane * 4;
  float m[G_], l[G_];
  float4 acc[G_];
#pragma unroll
  for (int g = 0; g < G_; ++g) {
    m[g] = MASKV; l[g] = 0.0f;
    acc[g].x = 0.0f; acc[g].y = 0.0f; acc[g].z = 0.0f; acc[g].w = 0.0f;
  }

  // ---- stream pages: wave wv handles pages wv, wv+8, ... (one page == one 16-token tile) ----
  for (int tt = wv; tt < P_; tt += WAVES_) {
    const int tbase = tt * TPP_;
    if (tbase >= seq_len) continue;  // fully masked page: contributes 0 weight (wave-uniform skip)

    const int page = page_map[b * P_ + tt];
    const float* kblk = key_pages   + ((size_t)kv * NUM_PAGES_ + page) * (TPP_ * D_);
    const float* vblk = value_pages + ((size_t)kv * NUM_PAGES_ + page) * (TPP_ * D_);

    // scores D[head, token] = Q(16x128) x K^T(128x16); fp16 inputs, fp32 accumulate
    v8f d = {};
    const float* krow = kblk + row16 * D_;  // B-matrix column = token row16
#pragma unroll
    for (int c = 0; c < 4; ++c) {
      const int base = c * 32 + hsel;
      v16h kf;
#pragma unroll
      for (int j = 0; j < 8; ++j) kf[j]     = (_Float16)krow[base + j];
#pragma unroll
      for (int j = 0; j < 8; ++j) kf[8 + j] = (_Float16)krow[base + 16 + j];
      d = __builtin_amdgcn_wmma_f32_16x16x32_f16(
              /*neg_a=*/false, qf[c], /*neg_b=*/false, kf,
              /*c_mod=*/(short)0, d, /*reuse_a=*/false, /*reuse_b=*/false);
    }

    // online softmax update per head; score for head g, token t sits in d[g] of lane t (<16)
    float w[G_];
#pragma unroll
    for (int g = 0; g < G_; ++g) {
      float s = d[g];
      const int tok = tbase + row16;
      if ((lane >= 16) || (tok >= seq_len)) s = MASKV;  // neutralize pad/garbage lanes
      const float tmax = wave_max32(s);
      const float mn   = fmaxf(m[g], tmax);
      const float cr   = __expf(m[g] - mn);
      w[g] = __expf(s - mn);                            // masked lanes underflow to 0
      l[g] = l[g] * cr + wave_sum32(w[g]);
      m[g] = mn;
      acc[g].x *= cr; acc[g].y *= cr; acc[g].z *= cr; acc[g].w *= cr;
    }

    // P.V: coalesced 512B V rows, weight broadcast from the owning lane
#pragma unroll
    for (int t = 0; t < TPP_; ++t) {
      const float4 vr = *(const float4*)(vblk + t * D_ + doff);
#pragma unroll
      for (int g = 0; g < G_; ++g) {
        const float wt = __shfl(w[g], t, 32);
        acc[g].x = fmaf(wt, vr.x, acc[g].x);
        acc[g].y = fmaf(wt, vr.y, acc[g].y);
        acc[g].z = fmaf(wt, vr.z, acc[g].z);
        acc[g].w = fmaf(wt, vr.w, acc[g].w);
      }
    }
  }

  // ---- cross-wave combine via LDS ----
#pragma unroll
  for (int g = 0; g < G_; ++g) {
    s_acc[wv][g][doff + 0] = acc[g].x;
    s_acc[wv][g][doff + 1] = acc[g].y;
    s_acc[wv][g][doff + 2] = acc[g].z;
    s_acc[wv][g][doff + 3] = acc[g].w;
  }
  if (lane == 0) {
#pragma unroll
    for (int g = 0; g < G_; ++g) { s_m[wv][g] = m[g]; s_l[wv][g] = l[g]; }
  }
  __syncthreads();

  if (wv < G_) {
    const int g = wv;
    float mg = MASKV;
#pragma unroll
    for (int w2 = 0; w2 < WAVES_; ++w2) mg = fmaxf(mg, s_m[w2][g]);
    float lg = 0.0f, ox = 0.0f, oy = 0.0f, oz = 0.0f, ow = 0.0f;
#pragma unroll
    for (int w2 = 0; w2 < WAVES_; ++w2) {
      const float e = __expf(s_m[w2][g] - mg);
      lg += s_l[w2][g] * e;
      ox = fmaf(e, s_acc[w2][g][doff + 0], ox);
      oy = fmaf(e, s_acc[w2][g][doff + 1], oy);
      oz = fmaf(e, s_acc[w2][g][doff + 2], oz);
      ow = fmaf(e, s_acc[w2][g][doff + 3], ow);
    }
    const float inv = 1.0f / lg;
    float* o = out + ((size_t)b * NQ_ + kv * G_ + g) * D_ + doff;
    o[0] = ox * inv; o[1] = oy * inv; o[2] = oz * inv; o[3] = ow * inv;
  }
}

extern "C" void kernel_launch(void* const* d_in, const int* in_sizes, int n_in,
                              void* d_out, int out_size, void* d_ws, size_t ws_size,
                              hipStream_t stream) {
  (void)in_sizes; (void)n_in; (void)out_size; (void)d_ws; (void)ws_size;
  const float* query       = (const float*)d_in[0];
  const float* key_pages   = (const float*)d_in[1];
  const float* value_pages = (const float*)d_in[2];
  const int*   page_map    = (const int*)d_in[3];
  const int*   seq_lengths = (const int*)d_in[4];
  float*       out         = (float*)d_out;

  dim3 grid(B_ * NKV_);     // 256 workgroups: one per (batch, kv-head)
  dim3 block(THREADS_);     // 8 wave32 waves
  hipLaunchKernelGGL(paged_attn_decode_kernel, grid, block, 0, stream,
                     query, key_pages, value_pages, page_map, seq_lengths, out);
}
